// VectorQuantization_68934225101058
// MI455X (gfx1250) — matrix operand order repaired and checked by reading, not compiled, not run on hardware
//
#include <hip/hip_runtime.h>
#include <hip/hip_bf16.h>

#define D_EMB   128
#define K_CODES 1024
#define N_VEC   65536          // B*T = 16*4096
#define DECAY   0.99f
#define EPS_    1e-5f
#define VQ_COMMIT 0.25f

typedef __attribute__((ext_vector_type(16))) __bf16       v16bf;
typedef __attribute__((ext_vector_type(8)))  float        v8f;
typedef __attribute__((ext_vector_type(4)))  unsigned int v4u;

// ---------------------------------------------------------------------------
// Fragment loaders from LDS (bf16, padded rows of 136 elements -> 16B aligned,
// conflict-free since 136*2B = 272B = 17-bank stride).
//
// A-matrix 16x32 bf16 layout (ISA 7.12.2): lane<16 holds row M=lane,
//   K = {0..7} then {16..23}; lane>=16 holds same row, K = {8..15},{24..31}.
// B-matrix 32x16 bf16 layout: lane<16 holds col N=lane, K = {0..7},{16..23};
//   lane>=16 holds col N=lane-16, K = {8..15},{24..31}.
// ---------------------------------------------------------------------------
__device__ __forceinline__ v16bf load_fragA(const __bf16* base, int t, int hi) {
  union { v4u u[2]; v16bf v; } x;
  x.u[0] = *(const v4u*)(base + t*32 + (hi ? 8  : 0));
  x.u[1] = *(const v4u*)(base + t*32 + (hi ? 24 : 16));
  return x.v;
}
__device__ __forceinline__ v16bf load_fragB(const __bf16* base, int t, int hi) {
  const int o = t*32 + hi*8;
  union { v4u u[2]; v16bf v; } x;
  x.u[0] = *(const v4u*)(base + o);
  x.u[1] = *(const v4u*)(base + o + 16);
  return x.v;
}

// ---------------------------------------------------------------------------
// Kernel 1: per-code squared norm, transposed codebook Et[K][D], zero accums.
// grid = K blocks of D threads.
// ---------------------------------------------------------------------------
__global__ void vq_prep(const float* __restrict__ E, float* __restrict__ enorm,
                        float* __restrict__ Et, float* __restrict__ embsumT,
                        float* __restrict__ counts, float* __restrict__ scal) {
  __shared__ float red[D_EMB];
  const int k = blockIdx.x, d = threadIdx.x;
  const float v = E[(size_t)d * K_CODES + k];
  Et[(size_t)k * D_EMB + d]      = v;
  embsumT[(size_t)k * D_EMB + d] = 0.f;
  red[d] = v * v;
  __syncthreads();
  for (int s = D_EMB / 2; s > 0; s >>= 1) {
    if (d < s) red[d] += red[d + s];
    __syncthreads();
  }
  if (d == 0) {
    enorm[k]  = red[0];
    counts[k] = 0.f;
    if (k == 0) { scal[0] = 0.f; scal[1] = 0.f; }   // sumsq, n
  }
}

// ---------------------------------------------------------------------------
// Kernel 2: fused bf16-WMMA distance GEMM + argmin, double-buffered B tile.
// Block = 256 threads (8 waves). Each wave owns 16 rows; block covers 128 rows.
// argmin over k of (||e_k||^2 - 2 x.e_k)  (||x||^2 constant per row).
// ---------------------------------------------------------------------------
__global__ __launch_bounds__(256)
void vq_argmin(const float* __restrict__ X, const float* __restrict__ E,
               const float* __restrict__ enorm,
               int* __restrict__ code_i, float* __restrict__ code_f) {
  __shared__ __align__(16) __bf16 Abuf[128][136];     // 128 rows of X, bf16
  __shared__ __align__(16) __bf16 Bbuf[2][16][136];   // double-buffered E tile

  const int tid    = threadIdx.x;
  const int wave   = tid >> 5;
  const int lane   = tid & 31;
  const int lane15 = lane & 15;
  const int hi     = lane >> 4;
  const int rowbase = blockIdx.x * 128;
  const int col    = tid & 15;          // fill mapping: col cycles over 16
  const int dbase  = tid >> 4;          // .. d advances by 16 per j-step

  // Stage X tile -> bf16 LDS (fully coalesced), and prologue-fill Bbuf[0]
  for (int j = 0; j < 64; ++j) {
    const int idx = tid + j * 256;                    // 0..16383
    const int r = idx >> 7, d = idx & 127;
    Abuf[r][d] = (__bf16)X[(size_t)(rowbase + r) * D_EMB + d];
  }
  #pragma unroll
  for (int j = 0; j < 8; ++j) {
    const int dd = dbase + j * 16;
    Bbuf[0][col][dd] = (__bf16)E[(size_t)dd * K_CODES + col];
  }
  __syncthreads();

  // Preload this wave's four A fragments (D = 4 * K32)
  const __bf16* arow = &Abuf[wave * 16 + lane15][0];
  v16bf afrag[4];
  #pragma unroll
  for (int t = 0; t < 4; ++t) afrag[t] = load_fragA(arow, t, hi);

  float minv[8]; int mink[8];
  #pragma unroll
  for (int i = 0; i < 8; ++i) { minv[i] = 3.0e38f; mink[i] = 0; }

  for (int c = 0; c < K_CODES / 16; ++c) {
    const int kb = c * 16;

    // Fill next chunk into the other buffer (overlaps with WMMA below).
    if (c + 1 < K_CODES / 16) {
      #pragma unroll
      for (int j = 0; j < 8; ++j) {
        const int dd = dbase + j * 16;
        Bbuf[(c + 1) & 1][col][dd] =
            (__bf16)E[(size_t)dd * K_CODES + kb + 16 + col];
      }
      if (kb + 32 < K_CODES)                         // warm chunk c+2
        __builtin_prefetch(&E[(size_t)(tid & 127) * K_CODES + kb + 32], 0, 1);
    }

    const float ec = enorm[kb + lane15];

    // All 8 ds_load_b128 issue before the WMMA chain (distinct registers).
    const __bf16* bptr = &Bbuf[c & 1][lane15][0];
    const v16bf b0 = load_fragB(bptr, 0, hi);
    const v16bf b1 = load_fragB(bptr, 1, hi);
    const v16bf b2 = load_fragB(bptr, 2, hi);
    const v16bf b3 = load_fragB(bptr, 3, hi);

    v8f acc = {};
    acc = __builtin_amdgcn_wmma_f32_16x16x32_bf16(false, afrag[0], false, b0,
                                                  (short)0, acc, false, false);
    acc = __builtin_amdgcn_wmma_f32_16x16x32_bf16(false, afrag[1], false, b1,
                                                  (short)0, acc, false, false);
    acc = __builtin_amdgcn_wmma_f32_16x16x32_bf16(false, afrag[2], false, b2,
                                                  (short)0, acc, false, false);
    acc = __builtin_amdgcn_wmma_f32_16x16x32_bf16(false, afrag[3], false, b3,
                                                  (short)0, acc, false, false);

    #pragma unroll
    for (int i = 0; i < 8; ++i) {
      const float val = ec - 2.0f * acc[i];
      if (val < minv[i]) { minv[i] = val; mink[i] = kb + lane15; }
    }

    __syncthreads();   // next-buffer fill done; current buffer free for reuse
  }

  // Cross-lane argmin within each 16-lane half (xor masks stay inside halves)
  #pragma unroll
  for (int i = 0; i < 8; ++i) {
    #pragma unroll
    for (int off = 8; off >= 1; off >>= 1) {
      const float ov = __shfl_xor(minv[i], off, 32);
      const int   ok = __shfl_xor(mink[i], off, 32);
      if (ov < minv[i] || (ov == minv[i] && ok < mink[i])) {
        minv[i] = ov; mink[i] = ok;
      }
    }
  }
  // C layout: lanes 0-15 -> rows i, lanes 16-31 -> rows i+8
  if (lane15 == 0) {
    #pragma unroll
    for (int i = 0; i < 8; ++i) {
      const int r = rowbase + wave * 16 + hi * 8 + i;
      code_i[r] = mink[i];
      code_f[r] = (float)mink[i];
    }
  }
}

// ---------------------------------------------------------------------------
// Kernel 3: codebook gather -> quantize, commit-loss partials, segment sums.
// Block = 256 threads = 8 rows, 32 lanes/row, 4 dims/lane.
// ---------------------------------------------------------------------------
__global__ __launch_bounds__(256)
void vq_gather(const float* __restrict__ X, const float* __restrict__ Et,
               const int* __restrict__ code_i,
               float* __restrict__ qout, float* __restrict__ embsumT,
               float* __restrict__ counts, float* __restrict__ sumsq) {
  const int wave = threadIdx.x >> 5, lane = threadIdx.x & 31;
  const int n = blockIdx.x * 8 + wave;
  const int k = code_i[n];
  const float* et = Et   + (size_t)k * D_EMB;
  const float* xr = X    + (size_t)n * D_EMB;
  float*       qr = qout + (size_t)n * D_EMB;

  float local = 0.f;
  #pragma unroll
  for (int j = 0; j < 4; ++j) {
    const int d = lane + j * 32;
    const float q = et[d], x = xr[d];
    qr[d] = q;                                      // quantize_st fwd == quantize
    const float df = q - x;
    local += df * df;
    atomicAdd(&embsumT[(size_t)k * D_EMB + d], x);  // segment_sum(flatten)
  }
  if (lane == 0) atomicAdd(&counts[k], 1.0f);       // segment_sum(ones)

  #pragma unroll
  for (int off = 16; off >= 1; off >>= 1) local += __shfl_down(local, off, 32);
  if (lane == 0) atomicAdd(sumsq, local);
}

// ---------------------------------------------------------------------------
// Kernel 4a: EMA cluster size + total n. Single block of K threads.
// ---------------------------------------------------------------------------
__global__ void vq_cluster(const float* __restrict__ cs_in,
                           const float* __restrict__ counts,
                           float* __restrict__ ncs_out, float* __restrict__ nsum) {
  __shared__ float red[K_CODES];
  const int k = threadIdx.x;
  const float v = cs_in[k] * DECAY + (1.f - DECAY) * counts[k];
  ncs_out[k] = v;
  red[k] = v;
  __syncthreads();
  for (int s = K_CODES / 2; s > 0; s >>= 1) {
    if (k < s) red[k] += red[k + s];
    __syncthreads();
  }
  if (k == 0) nsum[0] = red[0];
}

// ---------------------------------------------------------------------------
// Kernel 4b: EMA mean, normalized codebook, scalar diff. D*K elements.
// ---------------------------------------------------------------------------
__global__ __launch_bounds__(256)
void vq_final(const float* __restrict__ emean_in, const float* __restrict__ embsumT,
              const float* __restrict__ ncs, const float* __restrict__ nsum,
              const float* __restrict__ sumsq,
              float* __restrict__ emb_out, float* __restrict__ emean_out,
              float* __restrict__ diff_out) {
  const int idx = blockIdx.x * 256 + threadIdx.x;   // over [D][K] row-major
  const int d = idx >> 10, k = idx & (K_CODES - 1);
  const float es  = embsumT[(size_t)k * D_EMB + d];
  const float nem = emean_in[idx] * DECAY + (1.f - DECAY) * es;
  emean_out[idx] = nem;
  const float n  = nsum[0];
  const float cs = (ncs[k] + EPS_) / (n + (float)K_CODES * EPS_) * n;
  emb_out[idx] = nem / cs;
  if (idx == 0)
    diff_out[0] = VQ_COMMIT * sumsq[0] / (float)(N_VEC * D_EMB);
}

// ---------------------------------------------------------------------------
extern "C" void kernel_launch(void* const* d_in, const int* in_sizes, int n_in,
                              void* d_out, int out_size, void* d_ws, size_t ws_size,
                              hipStream_t stream) {
  (void)in_sizes; (void)n_in; (void)out_size; (void)ws_size;
  const float* X  = (const float*)d_in[0];   // [B,T,D]
  const float* E  = (const float*)d_in[1];   // [D,K]
  const float* CS = (const float*)d_in[2];   // [K]
  const float* EM = (const float*)d_in[3];   // [D,K]

  float* out      = (float*)d_out;
  float* q_out    = out;                         // N*D   = 8388608
  float* diff_o   = out + 8388608;               // 1
  float* code_f   = out + 8388609;               // N     = 65536
  float* emb_o    = out + 8454145;               // D*K   = 131072
  float* ncs_o    = out + 8585217;               // K     = 1024
  float* emean_o  = out + 8586241;               // D*K   = 131072  (total 8717313)

  float* ws       = (float*)d_ws;
  float* enorm    = ws;                          // K
  float* counts   = ws + 1024;                   // K
  float* scal     = ws + 2048;                   // [0]=sumsq, [1]=n (pad to 8)
  float* Et       = ws + 2056;                   // K*D (codebook transposed)
  float* embsumT  = Et + 131072;                 // K*D (segment sums, [K][D])
  int*   code_i   = (int*)(embsumT + 131072);    // N ints

  vq_prep  <<<K_CODES, D_EMB, 0, stream>>>(E, enorm, Et, embsumT, counts, scal);
  vq_argmin<<<N_VEC / 128, 256, 0, stream>>>(X, E, enorm, code_i, code_f);
  vq_gather<<<N_VEC / 8, 256, 0, stream>>>(X, Et, code_i, q_out, embsumT, counts,
                                           /*sumsq=*/scal);
  vq_cluster<<<1, K_CODES, 0, stream>>>(CS, counts, ncs_o, /*nsum=*/scal + 1);
  vq_final <<<(D_EMB * K_CODES) / 256, 256, 0, stream>>>(
      EM, embsumT, ncs_o, scal + 1, scal, emb_o, emean_o, diff_o);
}
